// ReSpropAttention_21895743275234
// MI455X (gfx1250) — matrix operand order repaired and checked by reading, not compiled, not run on hardware
//
#include <hip/hip_runtime.h>
#include <hip/hip_bf16.h>

// ---------- types ----------
typedef __bf16 v16bf __attribute__((ext_vector_type(16)));
typedef float  v8f   __attribute__((ext_vector_type(8)));
typedef unsigned int v4u __attribute__((ext_vector_type(4)));

#define B_DIM 8
#define S_DIM 2048
#define D_DIM 1024
#define LDT 40   // LDS row stride in bf16 elems (80B, 16B aligned, dodges bank conflicts)

// ---------- fragment helpers (wave32 WMMA layouts per CDNA5 ISA 7.12.2) ----------
__device__ __forceinline__ v16bf load_frag_a(const __bf16* p) {
  int lane = threadIdx.x & 31;
  int m = lane & 15, koff = (lane >> 4) * 8;
  const __bf16* r = p + m * LDT;
  v16bf f;
  ((v4u*)&f)[0] = *(const v4u*)(r + koff);
  ((v4u*)&f)[1] = *(const v4u*)(r + 16 + koff);
  return f;
}
__device__ __forceinline__ v16bf load_frag_b(const __bf16* p) {
  int lane = threadIdx.x & 31;
  int c = lane & 15, kh = lane >> 4;
  const __bf16* r = p + c * LDT + kh * 8;
  v16bf f;
  ((v4u*)&f)[0] = *(const v4u*)(r);
  ((v4u*)&f)[1] = *(const v4u*)(r + 16);
  return f;
}
__device__ __forceinline__ v8f wmma_bf16(v16bf a, v16bf b, v8f c) {
  return __builtin_amdgcn_wmma_f32_16x16x32_bf16(false, a, false, b, (short)0, c, false, false);
}
__device__ __forceinline__ v8f zero8() {
  v8f z;
  #pragma unroll
  for (int i = 0; i < 8; ++i) z[i] = 0.f;
  return z;
}

// ---------- async LDS fills (GLOBAL_LOAD_ASYNC_TO_LDS_B128, ASYNCcnt) ----------
// copies ROWS x 32 bf16 tile: one b128 chunk per lane per iteration, fully unrolled
template<int ROWS>
__device__ __forceinline__ void fill_async(__bf16* dst, const __bf16* src, long srcld) {
  #pragma unroll
  for (int i = 0; i < (ROWS * 4) / 256; ++i) {
    int c = threadIdx.x + i * 256;
    int row = c >> 2, off = (c & 3) * 8;
    const __bf16* g = src + (long)row * srcld + off;
    unsigned lds = (unsigned)(size_t)(dst + row * LDT + off);  // LDS byte offset (addr[31:0])
    asm volatile("global_load_async_to_lds_b128 %0, %1, off" :: "v"(lds), "v"(g) : "memory");
  }
}
__device__ __forceinline__ void wait_async0() {
  asm volatile("s_wait_asynccnt 0x0" ::: "memory");
}

// ---------- converting fill for f32 sources (K1), constant trip count ----------
template<int ROWS>
__device__ __forceinline__ void fill_f32t(__bf16* dst, const float* src, long srcld) {
  #pragma unroll
  for (int i = 0; i < (ROWS * 8) / 256; ++i) {
    int c = threadIdx.x + i * 256;
    int row = c >> 3, off = (c & 7) * 4;
    float4 f = *(const float4*)(src + (long)row * srcld + off);
    union { __bf16 h[4]; uint2 u; } pk;
    pk.h[0] = (__bf16)f.x; pk.h[1] = (__bf16)f.y; pk.h[2] = (__bf16)f.z; pk.h[3] = (__bf16)f.w;
    *(uint2*)(dst + row * LDT + off) = pk.u;
  }
}

// ---------- K1: QKV projection  out = X @ W^T + b (z selects q/k/v) ----------
__global__ __launch_bounds__(256) void k1_qkv(
    const float* __restrict__ X,
    const float* __restrict__ Wq, const float* __restrict__ bq,
    const float* __restrict__ Wk, const float* __restrict__ bk,
    const float* __restrict__ Wv, const float* __restrict__ bv,
    __bf16* __restrict__ Qb, __bf16* __restrict__ Kb, __bf16* __restrict__ Vt) {
  __shared__ __align__(16) __bf16 Xs[128 * LDT];
  __shared__ __align__(16) __bf16 Ws[64 * LDT];
  const int z = blockIdx.z;
  const float* W    = (z == 0) ? Wq : (z == 1) ? Wk : Wv;
  const float* bias = (z == 0) ? bq : (z == 1) ? bk : bv;
  const int n0 = blockIdx.x * 64;
  const long m0 = (long)blockIdx.y * 128;
  const int wave = threadIdx.x >> 5, lane = threadIdx.x & 31;
  const int col = lane & 15, mbase = (lane >> 4) * 8;

  v8f acc[4];
  #pragma unroll
  for (int t = 0; t < 4; ++t) {
    float bv_ = bias[n0 + t * 16 + col];
    #pragma unroll
    for (int v = 0; v < 8; ++v) acc[t][v] = bv_;
  }
  for (int k0 = 0; k0 < D_DIM; k0 += 32) {
    __syncthreads();
    fill_f32t<128>(Xs, X + m0 * D_DIM + k0, D_DIM);
    fill_f32t<64>(Ws, W + (long)n0 * D_DIM + k0, D_DIM);
    __syncthreads();
    v16bf a = load_frag_a(Xs + wave * 16 * LDT);
    v16bf bb[4];
    #pragma unroll
    for (int t = 0; t < 4; ++t) bb[t] = load_frag_b(Ws + t * 16 * LDT);
    #pragma unroll
    for (int t = 0; t < 4; ++t) acc[t] = wmma_bf16(a, bb[t], acc[t]);
  }
  if (z < 2) {
    __bf16* out = (z == 0) ? Qb : Kb;
    #pragma unroll
    for (int t = 0; t < 4; ++t)
      #pragma unroll
      for (int v = 0; v < 8; ++v)
        out[(m0 + wave * 16 + mbase + v) * D_DIM + (n0 + t * 16 + col)] = (__bf16)acc[t][v];
  } else {
    const long b = m0 / S_DIM;                       // 128-row tile never straddles a batch
    const long srow = (m0 % S_DIM) + wave * 16 + mbase;
    #pragma unroll
    for (int t = 0; t < 4; ++t) {
      union { __bf16 h[8]; v4u u; } pk;
      #pragma unroll
      for (int v = 0; v < 8; ++v) pk.h[v] = (__bf16)acc[t][v];
      *(v4u*)(Vt + (b * D_DIM + (n0 + t * 16 + col)) * S_DIM + srow) = pk.u;  // contiguous in s
    }
  }
}

// ---------- K2: scores = Q @ K^T / sqrt(D), tile 128(q) x 256(kv), async double-buffered ----------
__global__ __launch_bounds__(256) void k2_scores(
    const __bf16* __restrict__ Qb, const __bf16* __restrict__ Kb, float* __restrict__ Sc) {
  __shared__ __align__(16) __bf16 Qs[2][128 * LDT];   // 2 x 10 KiB
  __shared__ __align__(16) __bf16 Ks[2][256 * LDT];   // 2 x 20 KiB
  const int b = blockIdx.z;
  const int n0 = blockIdx.x * 256;
  const int m0 = blockIdx.y * 128;
  const __bf16* Qbase = Qb + ((long)b * S_DIM + m0) * D_DIM;
  const __bf16* Kbase = Kb + ((long)b * S_DIM + n0) * D_DIM;
  const int wave = threadIdx.x >> 5, lane = threadIdx.x & 31;
  v8f acc[16];
  #pragma unroll
  for (int t = 0; t < 16; ++t) acc[t] = zero8();

  fill_async<128>(Qs[0], Qbase, D_DIM);
  fill_async<256>(Ks[0], Kbase, D_DIM);
  wait_async0();
  __syncthreads();

  int cur = 0;
  for (int k0 = 0; k0 < D_DIM; k0 += 32) {
    const int nxt = cur ^ 1;
    if (k0 + 32 < D_DIM) {                       // prefetch next tile into other buffer
      fill_async<128>(Qs[nxt], Qbase + k0 + 32, D_DIM);
      fill_async<256>(Ks[nxt], Kbase + k0 + 32, D_DIM);
    }
    v16bf a = load_frag_a(Qs[cur] + wave * 16 * LDT);
    #pragma unroll
    for (int g = 0; g < 4; ++g) {
      v16bf bb[4];
      #pragma unroll
      for (int j = 0; j < 4; ++j) bb[j] = load_frag_b(Ks[cur] + (g * 4 + j) * 16 * LDT);
      #pragma unroll
      for (int j = 0; j < 4; ++j) acc[g * 4 + j] = wmma_bf16(a, bb[j], acc[g * 4 + j]);
    }
    wait_async0();
    __syncthreads();
    cur = nxt;
  }
  const float scale = 0.03125f;  // 1/sqrt(1024)
  const int col = lane & 15, mbase = (lane >> 4) * 8;
  const long qrow = (long)b * S_DIM + m0 + wave * 16 + mbase;
  #pragma unroll
  for (int t = 0; t < 16; ++t)
    #pragma unroll
    for (int v = 0; v < 8; ++v)
      Sc[(qrow + v) * S_DIM + (n0 + t * 16 + col)] = acc[t][v] * scale;
}

// ---------- K3: row softmax, f32 scores -> bf16 probs; one WG per row ----------
__global__ __launch_bounds__(256) void k3_softmax(const float* __restrict__ Sc, __bf16* __restrict__ P) {
  const long row = blockIdx.x;
  const float* src = Sc + row * S_DIM;
  __bf16* dst = P + row * S_DIM;
  __shared__ float redm[8], reds[8];
  float x[8];
  #pragma unroll
  for (int j = 0; j < 8; ++j) x[j] = src[threadIdx.x + j * 256];
  float m = -1e30f;
  #pragma unroll
  for (int j = 0; j < 8; ++j) m = fmaxf(m, x[j]);
  #pragma unroll
  for (int o = 16; o; o >>= 1) m = fmaxf(m, __shfl_xor(m, o, 32));
  if ((threadIdx.x & 31) == 0) redm[threadIdx.x >> 5] = m;
  __syncthreads();
  float gm = redm[0];
  #pragma unroll
  for (int w = 1; w < 8; ++w) gm = fmaxf(gm, redm[w]);
  float s = 0.f;
  #pragma unroll
  for (int j = 0; j < 8; ++j) { x[j] = __expf(x[j] - gm); s += x[j]; }
  #pragma unroll
  for (int o = 16; o; o >>= 1) s += __shfl_xor(s, o, 32);
  if ((threadIdx.x & 31) == 0) reds[threadIdx.x >> 5] = s;
  __syncthreads();
  float gs = reds[0];
  #pragma unroll
  for (int w = 1; w < 8; ++w) gs += reds[w];
  const float inv = 1.0f / gs;
  #pragma unroll
  for (int j = 0; j < 8; ++j) dst[threadIdx.x + j * 256] = (__bf16)(x[j] * inv);
}

// ---------- K4: out = P @ V, tile 128(q) x 64(e), async double-buffered over kv ----------
__global__ __launch_bounds__(256) void k4_pv(
    const __bf16* __restrict__ P, const __bf16* __restrict__ Vt, float* __restrict__ Out) {
  __shared__ __align__(16) __bf16 Ps[2][128 * LDT];
  __shared__ __align__(16) __bf16 Vs[2][64 * LDT];
  const int b = blockIdx.z;
  const int n0 = blockIdx.x * 64;   // e
  const int m0 = blockIdx.y * 128;  // q
  const __bf16* Pbase = P + ((long)b * S_DIM + m0) * S_DIM;
  const __bf16* Vbase = Vt + ((long)b * D_DIM + n0) * S_DIM;
  const int wave = threadIdx.x >> 5, lane = threadIdx.x & 31;
  v8f acc[4];
  #pragma unroll
  for (int t = 0; t < 4; ++t) acc[t] = zero8();

  fill_async<128>(Ps[0], Pbase, S_DIM);
  fill_async<64>(Vs[0], Vbase, S_DIM);
  wait_async0();
  __syncthreads();

  int cur = 0;
  for (int k0 = 0; k0 < S_DIM; k0 += 32) {
    const int nxt = cur ^ 1;
    if (k0 + 32 < S_DIM) {
      fill_async<128>(Ps[nxt], Pbase + k0 + 32, S_DIM);
      fill_async<64>(Vs[nxt], Vbase + k0 + 32, S_DIM);
    }
    v16bf a = load_frag_a(Ps[cur] + wave * 16 * LDT);
    v16bf bb[4];
    #pragma unroll
    for (int t = 0; t < 4; ++t) bb[t] = load_frag_b(Vs[cur] + t * 16 * LDT);
    #pragma unroll
    for (int t = 0; t < 4; ++t) acc[t] = wmma_bf16(a, bb[t], acc[t]);
    wait_async0();
    __syncthreads();
    cur = nxt;
  }
  const int col = lane & 15, mbase = (lane >> 4) * 8;
  const long qrow = (long)b * S_DIM + m0 + wave * 16 + mbase;
  #pragma unroll
  for (int t = 0; t < 4; ++t)
    #pragma unroll
    for (int v = 0; v < 8; ++v)
      Out[(qrow + v) * D_DIM + (n0 + t * 16 + col)] = acc[t][v];
}

// ---------- host launcher ----------
extern "C" void kernel_launch(void* const* d_in, const int* in_sizes, int n_in,
                              void* d_out, int out_size, void* d_ws, size_t ws_size,
                              hipStream_t stream) {
  (void)in_sizes; (void)n_in; (void)out_size; (void)ws_size;
  const float* X  = (const float*)d_in[0];
  const float* Wq = (const float*)d_in[1];
  const float* bq = (const float*)d_in[2];
  const float* Wk = (const float*)d_in[3];
  const float* bk = (const float*)d_in[4];
  const float* Wv = (const float*)d_in[5];
  const float* bv = (const float*)d_in[6];
  float* out = (float*)d_out;

  char* ws = (char*)d_ws;
  const size_t QKV_BYTES = (size_t)B_DIM * S_DIM * D_DIM * 2;       // 32 MiB each (bf16)
  const size_t SC_BYTES  = (size_t)B_DIM * S_DIM * S_DIM * 4;       // 128 MiB (f32)
  __bf16* Qb = (__bf16*)(ws);
  __bf16* Kb = (__bf16*)(ws + QKV_BYTES);
  __bf16* Vt = (__bf16*)(ws + 2 * QKV_BYTES);
  float*  Sc = (float*)(ws + 3 * QKV_BYTES);
  __bf16* Pp = (__bf16*)(ws + 3 * QKV_BYTES + SC_BYTES);

  dim3 blk(256);
  k1_qkv<<<dim3(D_DIM / 64, (B_DIM * S_DIM) / 128, 3), blk, 0, stream>>>(
      X, Wq, bq, Wk, bk, Wv, bv, Qb, Kb, Vt);
  k2_scores<<<dim3(S_DIM / 256, S_DIM / 128, B_DIM), blk, 0, stream>>>(Qb, Kb, Sc);
  k3_softmax<<<dim3(B_DIM * S_DIM), blk, 0, stream>>>(Sc, Pp);
  k4_pv<<<dim3(D_DIM / 64, S_DIM / 128, B_DIM), blk, 0, stream>>>(Pp, Vt, out);
}
